// AEDiscriminator_63299228009393
// MI455X (gfx1250) — matrix operand order repaired and checked by reading, not compiled, not run on hardware
//
#include <hip/hip_runtime.h>
#include <hip/hip_bf16.h>

// ---------------------------------------------------------------------------
// CDNA5 (gfx1250) persistent RNN stack: 2x GRU + 2x LSTM, masked, fused head.
// bf16 WMMA (v_wmma_f32_16x16x32_bf16) with fp32 accumulate. 8 resident WGs
// (4 per stack) sync via global barrier per layer. Weights AND the input x are
// pre-converted/packed to bf16 once per call, so the 512-step recurrent loop
// does only b128 loads + WMMA + gate math.
// ---------------------------------------------------------------------------

typedef __attribute__((ext_vector_type(16))) __bf16 v16bf;
typedef __attribute__((ext_vector_type(8)))  __bf16 v8bf;
typedef __attribute__((ext_vector_type(8)))  float  v8f;

#define B_   64
#define T_   512
#define D_   96
#define H_   256
#define XROWSTRIDE (T_ * D_)     // 49152 (elements)
#define STATE_SZ   (B_ * H_)     // 16384 elements per state buffer

__device__ __forceinline__ float sigm(float x) { return 1.0f / (1.0f + __expf(-x)); }
__device__ __forceinline__ float tanh_(float x) {
    float xc = fminf(fmaxf(x, -15.0f), 15.0f);
    float e = __expf(2.0f * xc);
    return (e - 1.0f) / (e + 1.0f);
}

__device__ __forceinline__ v8f wmma_bf(v16bf a, v16bf b, v8f c) {
    return __builtin_amdgcn_wmma_f32_16x16x32_bf16(false, a, false, b, (short)0, c, false, false);
}

// Packed B tile: 32 lanes x 16 bf16, contiguous; lane reads 32 bytes.
__device__ __forceinline__ v16bf load_packedB(const __bf16* base, int lane) {
    const v8bf* p = (const v8bf*)(base + lane * 16);
    v8bf lo = p[0], hi = p[1];
    v16bf r;
#pragma unroll
    for (int i = 0; i < 8; ++i) { r[i] = lo[i]; r[i + 8] = hi[i]; }
    return r;
}

// A operand from bf16 row-major [rows][ld]: row = m0+(lane&15),
// elements 0..7 at K=k0+kh, elements 8..15 at K=k0+16+kh, kh=(lane>>4)*8.
__device__ __forceinline__ v16bf loadA_bf(const __bf16* m, int ld, int m0, int k0, int lane) {
    int row = m0 + (lane & 15);
    int kh  = k0 + ((lane >> 4) << 3);
    const __bf16* p = m + (size_t)row * ld + kh;
    v8bf a = *(const v8bf*)p;
    v8bf b = *(const v8bf*)(p + 16);
    v16bf r;
#pragma unroll
    for (int i = 0; i < 8; ++i) { r[i] = a[i]; r[i + 8] = b[i]; }
    return r;
}

// Global barrier across the 4 workgroups of one stack group.
__device__ __forceinline__ void gbarrier(int* cnt, int target) {
    __syncthreads();
    if (threadIdx.x == 0) {
        __threadfence();
        atomicAdd(cnt, 1);
        while (atomicAdd(cnt, 0) < target) __builtin_amdgcn_s_sleep(1);
    }
    __syncthreads();
    __threadfence();
}

// ------------------------- GRU layer tile (16x16 of h) ----------------------
// KWT = k-tiles of the input-projection GEMM (3 for x with D=96, 8 for h).
template <int KWT>
__device__ void gru_layer(int tile, int lane, int t,
                          const __bf16* aMat, int aLd,
                          const __bf16* Wp, const __bf16* Up,
                          const float* bi, const float* br,
                          const float* hcur, const __bf16* hcurb,
                          float* hnext, __bf16* hnextb,
                          const int* lengths, const float* outw, float* part) {
    const int mi = tile & 3, jt = tile >> 2;
    const int m0 = mi << 4, j0 = jt << 4;
    v8f Az{}, Ar{}, Axh{}, Arh{};
#pragma unroll
    for (int kt = 0; kt < KWT; ++kt) {
        v16bf A = loadA_bf(aMat, aLd, m0, kt * 32, lane);
        Az  = wmma_bf(A, load_packedB(Wp + ((size_t)(jt     ) * KWT + kt) * 512, lane), Az);
        Ar  = wmma_bf(A, load_packedB(Wp + ((size_t)(jt + 16) * KWT + kt) * 512, lane), Ar);
        Axh = wmma_bf(A, load_packedB(Wp + ((size_t)(jt + 32) * KWT + kt) * 512, lane), Axh);
    }
#pragma unroll
    for (int kt = 0; kt < 8; ++kt) {   // recurrent GEMM, K = H = 256
        v16bf A = loadA_bf(hcurb, H_, m0, kt * 32, lane);
        Az  = wmma_bf(A, load_packedB(Up + ((size_t)(jt     ) * 8 + kt) * 512, lane), Az);
        Ar  = wmma_bf(A, load_packedB(Up + ((size_t)(jt + 16) * 8 + kt) * 512, lane), Ar);
        Arh = wmma_bf(A, load_packedB(Up + ((size_t)(jt + 32) * 8 + kt) * 512, lane), Arh);
    }
    const int col = lane & 15, rb = (lane >> 4) << 3;
    const int n = j0 + col;
    const float bz  = bi[n]        + br[n];
    const float brg = bi[n + 256]  + br[n + 256];
    const float bxh = bi[n + 512];
    const float brh = br[n + 512];
    const float w = outw ? outw[n] : 0.0f;
    float ps[8];
#pragma unroll
    for (int e = 0; e < 8; ++e) {
        const int m = m0 + rb + e;                       // batch row
        const float hold = hcur[m * H_ + n];
        const float z  = sigm(Az[e] + bz);
        const float r  = sigm(Ar[e] + brg);
        const float hh = tanh_(Axh[e] + bxh + r * (Arh[e] + brh));
        float hn = z * hold + (1.0f - z) * hh;
        hn = (t < lengths[m]) ? hn : hold;               // freeze under mask
        hnext[m * H_ + n]  = hn;
        hnextb[m * H_ + n] = (__bf16)hn;                 // hw cvt
        ps[e] = hn * w;
    }
    if (outw) {                                          // fused head partial
#pragma unroll
        for (int e = 0; e < 8; ++e) {
            float s = ps[e];
            s += __shfl_xor(s, 1, 32); s += __shfl_xor(s, 2, 32);
            s += __shfl_xor(s, 4, 32); s += __shfl_xor(s, 8, 32);
            if (col == 0) part[((size_t)t * 16 + jt) * 64 + (m0 + rb + e)] = s;
        }
    }
}

// ------------------------- LSTM layer tile ---------------------------------
template <int KWT>
__device__ void lstm_layer(int tile, int lane, int t,
                           const __bf16* aMat, int aLd,
                           const __bf16* Wp, const __bf16* Up, const float* b,
                           const float* hcur, const __bf16* hcurb, const float* ccur,
                           float* hnext, __bf16* hnextb, float* cnext,
                           const int* lengths, const float* outw, float* part) {
    const int mi = tile & 3, jt = tile >> 2;
    const int m0 = mi << 4, j0 = jt << 4;
    v8f Ai{}, Af{}, Ac{}, Ao{};
#pragma unroll
    for (int kt = 0; kt < KWT; ++kt) {
        v16bf A = loadA_bf(aMat, aLd, m0, kt * 32, lane);
        Ai = wmma_bf(A, load_packedB(Wp + ((size_t)(jt     ) * KWT + kt) * 512, lane), Ai);
        Af = wmma_bf(A, load_packedB(Wp + ((size_t)(jt + 16) * KWT + kt) * 512, lane), Af);
        Ac = wmma_bf(A, load_packedB(Wp + ((size_t)(jt + 32) * KWT + kt) * 512, lane), Ac);
        Ao = wmma_bf(A, load_packedB(Wp + ((size_t)(jt + 48) * KWT + kt) * 512, lane), Ao);
    }
#pragma unroll
    for (int kt = 0; kt < 8; ++kt) {
        v16bf A = loadA_bf(hcurb, H_, m0, kt * 32, lane);
        Ai = wmma_bf(A, load_packedB(Up + ((size_t)(jt     ) * 8 + kt) * 512, lane), Ai);
        Af = wmma_bf(A, load_packedB(Up + ((size_t)(jt + 16) * 8 + kt) * 512, lane), Af);
        Ac = wmma_bf(A, load_packedB(Up + ((size_t)(jt + 32) * 8 + kt) * 512, lane), Ac);
        Ao = wmma_bf(A, load_packedB(Up + ((size_t)(jt + 48) * 8 + kt) * 512, lane), Ao);
    }
    const int col = lane & 15, rb = (lane >> 4) << 3;
    const int n = j0 + col;
    const float bi_ = b[n], bf_ = b[n + 256], bc_ = b[n + 512], bo_ = b[n + 768];
    const float w = outw ? outw[n] : 0.0f;
    float ps[8];
#pragma unroll
    for (int e = 0; e < 8; ++e) {
        const int m = m0 + rb + e;
        const float hold = hcur[m * H_ + n];
        const float cold = ccur[m * H_ + n];
        const float ig = sigm(Ai[e] + bi_);
        const float fg = sigm(Af[e] + bf_);
        const float cg = tanh_(Ac[e] + bc_);
        const float og = sigm(Ao[e] + bo_);
        float cn = fg * cold + ig * cg;
        float hn = og * tanh_(cn);
        const bool mk = (t < lengths[m]);
        hn = mk ? hn : hold;
        cn = mk ? cn : cold;
        hnext[m * H_ + n]  = hn;
        hnextb[m * H_ + n] = (__bf16)hn;
        cnext[m * H_ + n]  = cn;
        ps[e] = hn * w;
    }
    if (outw) {
#pragma unroll
        for (int e = 0; e < 8; ++e) {
            float s = ps[e];
            s += __shfl_xor(s, 1, 32); s += __shfl_xor(s, 2, 32);
            s += __shfl_xor(s, 4, 32); s += __shfl_xor(s, 8, 32);
            if (col == 0) part[((size_t)t * 16 + jt) * 64 + (m0 + rb + e)] = s;
        }
    }
}

// ------------------------- persistent main kernel --------------------------
__global__ __launch_bounds__(256, 1)
void rnn_kernel(const __bf16* __restrict__ xb, const int* __restrict__ lengths,
                const float* g_bi0, const float* g_br0,
                const float* g_bi1, const float* g_br1,
                const float* l_b0, const float* l_b1,
                const float* out_W,
                const __bf16* gW0p, const __bf16* gU0p,
                const __bf16* gW1p, const __bf16* gU1p,
                const __bf16* lW0p, const __bf16* lU0p,
                const __bf16* lW1p, const __bf16* lU1p,
                float* hg0, float* hg1, float* hl0, float* hl1,
                float* cl0, float* cl1,
                __bf16* hg0b, __bf16* hg1b, __bf16* hl0b, __bf16* hl1b,
                float* partG, float* partL, int* bars) {
    const bool isL = blockIdx.x >= 4;
    int* cnt = bars + (isL ? 64 : 0);                 // separate cachelines
    const int gw   = ((int)blockIdx.x & 3) * 8 + ((int)threadIdx.x >> 5); // 0..31
    const int lane = (int)threadIdx.x & 31;
    int target = 0;
    const int SZ = STATE_SZ;

    for (int t = 0; t < T_; ++t) {
        const int p = t & 1, q = p ^ 1;
        const __bf16* xt = xb + (size_t)t * D_;       // row stride = T*D
        if (!isL) {
            for (int tile = gw; tile < 64; tile += 32)
                gru_layer<3>(tile, lane, t, xt, XROWSTRIDE, gW0p, gU0p, g_bi0, g_br0,
                             hg0 + p * SZ, hg0b + p * SZ,
                             hg0 + q * SZ, hg0b + q * SZ,
                             lengths, nullptr, nullptr);
            target += 4; gbarrier(cnt, target);
            for (int tile = gw; tile < 64; tile += 32)
                gru_layer<8>(tile, lane, t, hg0b + q * SZ, H_, gW1p, gU1p,
                             g_bi1, g_br1,
                             hg1 + p * SZ, hg1b + p * SZ,
                             hg1 + q * SZ, hg1b + q * SZ,
                             lengths, out_W, partG);
            target += 4; gbarrier(cnt, target);
        } else {
            for (int tile = gw; tile < 64; tile += 32)
                lstm_layer<3>(tile, lane, t, xt, XROWSTRIDE, lW0p, lU0p, l_b0,
                              hl0 + p * SZ, hl0b + p * SZ, cl0 + p * SZ,
                              hl0 + q * SZ, hl0b + q * SZ, cl0 + q * SZ,
                              lengths, nullptr, nullptr);
            target += 4; gbarrier(cnt, target);
            for (int tile = gw; tile < 64; tile += 32)
                lstm_layer<8>(tile, lane, t, hl0b + q * SZ, H_, lW1p, lU1p, l_b1,
                              hl1 + p * SZ, hl1b + p * SZ, cl1 + p * SZ,
                              hl1 + q * SZ, hl1b + q * SZ, cl1 + q * SZ,
                              lengths, out_W + 256, partL);
            target += 4; gbarrier(cnt, target);
        }
    }
}

// ------------------------- init / pack / head kernels ----------------------
__global__ void zero_ws(int* p, int n) {
    int i = blockIdx.x * blockDim.x + threadIdx.x;
    if (i < n) p[i] = 0;
}

// f32 -> bf16 elementwise (for x pre-conversion).
__global__ void cvt_bf16(const float* __restrict__ src, __bf16* __restrict__ dst, int n) {
    for (int i = blockIdx.x * blockDim.x + threadIdx.x; i < n;
         i += gridDim.x * blockDim.x)
        dst[i] = (__bf16)src[i];
}

// Pack f32 weight [K][N] -> bf16 WMMA B-operand tiles.
// tile = nt*ktiles + kt ; within tile: lane(32) x j(16);
// n = nt*16 + (lane&15), k = kt*32 + (lane>>4)*8 + (j&7) + (j>>3)*16.
__global__ void pack_weights(const float* __restrict__ src, __bf16* __restrict__ dst,
                             int K, int N) {
    int ktiles = K >> 5;
    int total = (K >> 5) * (N >> 4) * 512;
    for (int i = blockIdx.x * blockDim.x + threadIdx.x; i < total;
         i += gridDim.x * blockDim.x) {
        int j    = i & 15;
        int lane = (i >> 4) & 31;
        int tile = i >> 9;
        int kt = tile % ktiles;
        int nt = tile / ktiles;
        int n = (nt << 4) + (lane & 15);
        int k = (kt << 5) + ((lane >> 4) << 3) + (j & 7) + ((j >> 3) << 4);
        dst[i] = (__bf16)src[(size_t)k * N + n];
    }
}

__global__ void head_kernel(const float* __restrict__ partG,
                            const float* __restrict__ partL,
                            const float* __restrict__ out_b,
                            float* __restrict__ out) {
    int i = blockIdx.x * blockDim.x + threadIdx.x;   // i = b*T + t
    if (i >= B_ * T_) return;
    int b = i / T_, t = i % T_;
    float s = out_b[0];
#pragma unroll
    for (int jt = 0; jt < 16; ++jt) {
        s += partG[((size_t)t * 16 + jt) * 64 + b];
        s += partL[((size_t)t * 16 + jt) * 64 + b];
    }
    out[i] = s;
}

// ---------------------------------------------------------------------------
extern "C" void kernel_launch(void* const* d_in, const int* in_sizes, int n_in,
                              void* d_out, int out_size, void* d_ws, size_t ws_size,
                              hipStream_t stream) {
    (void)in_sizes; (void)n_in; (void)out_size; (void)ws_size;
    const float* x       = (const float*)d_in[0];
    const int*   lengths = (const int*)  d_in[1];
    const float* g_W0 = (const float*)d_in[2];
    const float* g_U0 = (const float*)d_in[3];
    const float* g_bi0 = (const float*)d_in[4];
    const float* g_br0 = (const float*)d_in[5];
    const float* g_W1 = (const float*)d_in[6];
    const float* g_U1 = (const float*)d_in[7];
    const float* g_bi1 = (const float*)d_in[8];
    const float* g_br1 = (const float*)d_in[9];
    const float* l_W0 = (const float*)d_in[10];
    const float* l_U0 = (const float*)d_in[11];
    const float* l_b0 = (const float*)d_in[12];
    const float* l_W1 = (const float*)d_in[13];
    const float* l_U1 = (const float*)d_in[14];
    const float* l_b1 = (const float*)d_in[15];
    const float* out_W = (const float*)d_in[16];
    const float* out_b = (const float*)d_in[17];

    char* ws = (char*)d_ws;
    size_t off = 0;
    auto alloc = [&](size_t bytes) -> void* {
        off = (off + 255) & ~(size_t)255;
        void* p = ws + off;
        off += bytes;
        return p;
    };

    int* bars = (int*)alloc(512);
    float* hg0 = (float*)alloc(2 * STATE_SZ * sizeof(float));
    float* hg1 = (float*)alloc(2 * STATE_SZ * sizeof(float));
    float* hl0 = (float*)alloc(2 * STATE_SZ * sizeof(float));
    float* hl1 = (float*)alloc(2 * STATE_SZ * sizeof(float));
    float* cl0 = (float*)alloc(2 * STATE_SZ * sizeof(float));
    float* cl1 = (float*)alloc(2 * STATE_SZ * sizeof(float));
    __bf16* hg0b = (__bf16*)alloc(2 * STATE_SZ * sizeof(__bf16));
    __bf16* hg1b = (__bf16*)alloc(2 * STATE_SZ * sizeof(__bf16));
    __bf16* hl0b = (__bf16*)alloc(2 * STATE_SZ * sizeof(__bf16));
    __bf16* hl1b = (__bf16*)alloc(2 * STATE_SZ * sizeof(__bf16));
    size_t zeroEnd = (off + 3) & ~(size_t)3;   // bytes to zero from ws start

    auto wbytes = [](int K, int N) { return (size_t)(K >> 5) * (N >> 4) * 512 * sizeof(__bf16); };
    __bf16* gW0p = (__bf16*)alloc(wbytes(96, 768));
    __bf16* gU0p = (__bf16*)alloc(wbytes(256, 768));
    __bf16* gW1p = (__bf16*)alloc(wbytes(256, 768));
    __bf16* gU1p = (__bf16*)alloc(wbytes(256, 768));
    __bf16* lW0p = (__bf16*)alloc(wbytes(96, 1024));
    __bf16* lU0p = (__bf16*)alloc(wbytes(256, 1024));
    __bf16* lW1p = (__bf16*)alloc(wbytes(256, 1024));
    __bf16* lU1p = (__bf16*)alloc(wbytes(256, 1024));
    __bf16* xb   = (__bf16*)alloc((size_t)B_ * T_ * D_ * sizeof(__bf16));
    float* partG = (float*)alloc((size_t)T_ * 16 * 64 * sizeof(float));
    float* partL = (float*)alloc((size_t)T_ * 16 * 64 * sizeof(float));

    // 1) zero barriers + state double-buffers (fp32 and bf16)
    {
        int nInts = (int)(zeroEnd / 4);
        zero_ws<<<(nInts + 255) / 256, 256, 0, stream>>>((int*)d_ws, nInts);
    }
    // 2) pre-convert x to bf16; pack weights into WMMA B-operand layout
    {
        int n = B_ * T_ * D_;
        cvt_bf16<<<(n + 255) / 256, 256, 0, stream>>>(x, xb, n);
    }
    auto pack = [&](const float* s, __bf16* d, int K, int N) {
        int total = (K >> 5) * (N >> 4) * 512;
        pack_weights<<<(total + 255) / 256, 256, 0, stream>>>(s, d, K, N);
    };
    pack(g_W0, gW0p, 96, 768);
    pack(g_U0, gU0p, 256, 768);
    pack(g_W1, gW1p, 256, 768);
    pack(g_U1, gU1p, 256, 768);
    pack(l_W0, lW0p, 96, 1024);
    pack(l_U0, lU0p, 256, 1024);
    pack(l_W1, lW1p, 256, 1024);
    pack(l_U1, lU1p, 256, 1024);

    // 3) persistent recurrent kernel: 8 WGs (4 GRU + 4 LSTM), 8 waves each
    rnn_kernel<<<8, 256, 0, stream>>>(
        xb, lengths, g_bi0, g_br0, g_bi1, g_br1, l_b0, l_b1, out_W,
        gW0p, gU0p, gW1p, gU1p, lW0p, lU0p, lW1p, lU1p,
        hg0, hg1, hl0, hl1, cl0, cl1,
        hg0b, hg1b, hl0b, hl1b,
        partG, partL, bars);

    // 4) deterministic head reduction -> d_out [B,T,1]
    head_kernel<<<(B_ * T_ + 255) / 256, 256, 0, stream>>>(
        partG, partL, out_b, (float*)d_out);
}
